// TurboQuantMSE_61701500174504
// MI455X (gfx1250) — matrix operand order repaired and checked by reading, compiled, not run on hardware
//
#include <hip/hip_runtime.h>
#include <hip/hip_bf16.h>
#include <stdint.h>

typedef __attribute__((ext_vector_type(16))) _Float16 v16h;
typedef __attribute__((ext_vector_type(4)))  _Float16 v4h;
typedef __attribute__((ext_vector_type(8)))  float    v8f;

#define D 128          // head dim
#define NTILES 8       // 128 / 16 output-column tiles
#define KSTEPS 4       // 128 / 32 K chunks per WMMA chain
#define WAVES 4        // waves per block
#define ROWS_PER_WAVE 32

union FragU {
    uint4 u[2];
    v16h  h;
};

// ---------------------------------------------------------------------------
// Prep: build f16 B-matrix WMMA fragments for B1 = Pi^T (rotate) and
// B2 = Pi (un-rotate), pre-swizzled to a wave32 B layout in contiguous k order (same as A):
//   fragment (ks, nt): lane l (0..15) holds column n = nt*16 + l, K = k0..k0+15
//                      lane l (16..31) holds column n = nt*16 + (l-16), K = k0+16..k0+31
//   dword v (0..7) packs halves (K = base + 2v, base + 2v + 1)
// Storage: frag[ks][nt][lane][v]  (lane's 8 dwords contiguous -> 2x b128 loads)
// ---------------------------------------------------------------------------
__global__ void tq_prep_frags(const float* __restrict__ Pi,
                              uint32_t* __restrict__ b1,
                              uint32_t* __restrict__ b2) {
    int t = blockIdx.x * blockDim.x + threadIdx.x;   // 0 .. 8191
    if (t >= KSTEPS * NTILES * 32 * 8) return;
    int v  = t & 7;
    int l  = (t >> 3) & 31;
    int nt = (t >> 8) & 7;
    int ks = t >> 11;

    int n = nt * 16 + (l & 15);
    int k = ks * 32 + ((l >> 4) << 4) + 2 * v;

    union { _Float16 h[2]; uint32_t u; } p;

    // B1[k][n] = Pi[n][k]   (rotated = x_unit @ Pi^T)
    p.h[0] = (_Float16)Pi[n * D + k];
    p.h[1] = (_Float16)Pi[n * D + k + 1];
    b1[t] = p.u;

    // B2[k][n] = Pi[k][n]   (recon = values @ Pi)
    p.h[0] = (_Float16)Pi[k * D + n];
    p.h[1] = (_Float16)Pi[(k + 1) * D + n];
    b2[t] = p.u;
}

// ---------------------------------------------------------------------------
// Main kernel: one wave fully processes 32 rows (two 16-row WMMA chunks),
// so each B fragment load feeds two WMMAs (halves LDS/L2 B traffic).
// ---------------------------------------------------------------------------
__global__ __launch_bounds__(WAVES * 32) void tq_main(const float* __restrict__ x,
                                                      const float* __restrict__ cen,
                                                      const float* __restrict__ bnd,
                                                      const uint32_t* __restrict__ b1g,
                                                      const uint32_t* __restrict__ b2g,
                                                      float* __restrict__ out,
                                                      int nrows) {
    __shared__ uint32_t  sB1[KSTEPS * NTILES * 32 * 8];   // 32 KB: B1 fragments
    __shared__ _Float16  sX[WAVES][ROWS_PER_WAVE * D];    // 32 KB: per-wave scratch

    const int tid  = threadIdx.x;
    const int lane = tid & 31;
    const int w    = tid >> 5;
    const int rlo  = lane & 15;          // row-in-chunk owned by this lane (A layout)
    const int hsel = lane >> 4;          // half-wave select

    // Stage B1 fragments to LDS (whole block cooperates, 16x uint4 per thread).
    {
        const uint4* src = (const uint4*)b1g;
        uint4*       dst = (uint4*)sB1;
#pragma unroll
        for (int i = 0; i < 16; ++i) dst[i * 128 + tid] = src[i * 128 + tid];
    }
    __syncthreads();

    // Quantizer constants (symmetric Lloyd-Max codebook): upper-half interior
    // boundaries bnd[9..15] and centroids cen[8..15]; sign restored via bit xor.
    float qb[7], qc[8];
#pragma unroll
    for (int i = 0; i < 7; ++i) qb[i] = bnd[9 + i];
#pragma unroll
    for (int i = 0; i < 8; ++i) qc[i] = cen[8 + i];

    const long row0 = ((long)blockIdx.x * WAVES + w) * ROWS_PER_WAVE;
    if (row0 >= nrows) return;
    const float* xr   = x + row0 * D;
    float*       outr = out + row0 * D;
    _Float16*    myS  = sX[w];

    // ---- Pass 1: load 32 rows (512B coalesced per instruction), per-row norms
    // via shuffle butterflies, write x_unit as f16 row-major into LDS. ----
    float nq0 = 0.0f, nq1 = 0.0f;   // fp16-rounded norms of rows rlo, 16+rlo
#pragma unroll
    for (int j = 0; j < ROWS_PER_WAVE; ++j) {
        float4 xv = ((const float4*)(xr + j * D))[lane];
        float s = xv.x * xv.x + xv.y * xv.y + xv.z * xv.z + xv.w * xv.w;
        s += __shfl_xor(s, 1);
        s += __shfl_xor(s, 2);
        s += __shfl_xor(s, 4);
        s += __shfl_xor(s, 8);
        s += __shfl_xor(s, 16);
        float nrm = __builtin_sqrtf(s);
        float nqj = (float)(_Float16)nrm;   // fp16 round-trip (norm_dtype=float16)
        if (rlo == (j & 15)) {
            if (j < 16) nq0 = nqj; else nq1 = nqj;
        }
        float inv = 1.0f / (nrm + 1e-8f);
        v4h hp;
        hp.x = (_Float16)(xv.x * inv);
        hp.y = (_Float16)(xv.y * inv);
        hp.z = (_Float16)(xv.z * inv);
        hp.w = (_Float16)(xv.w * inv);
        *(v4h*)(myS + j * D + lane * 4) = hp;
    }

    __asm__ __volatile__("s_wait_dscnt 0" ::: "memory");  // LDS writes -> reads, same wave

    // ---- A1 fragments (x_unit) in contiguous k order (matching B frags), both 16-row chunks:
    // lane holds row rlo (+16 for chunk1); dwords0-3: K = k0+16*hsel.. ; dwords4-7: +8. ----
    v16h a1[2][KSTEPS];
    const int hbase = rlo * D + 16 * hsel;
#pragma unroll
    for (int c = 0; c < 2; ++c) {
#pragma unroll
        for (int ks = 0; ks < KSTEPS; ++ks) {
            FragU f;
            f.u[0] = *(const uint4*)(myS + c * 16 * D + hbase + ks * 32);
            f.u[1] = *(const uint4*)(myS + c * 16 * D + hbase + ks * 32 + 8);
            a1[c][ks] = f.h;
        }
    }

    __asm__ __volatile__("s_wait_dscnt 0" ::: "memory");  // A1 in regs before scratch reuse

    // ---- GEMM1 (rotate) + quantize; each B frag load feeds 2 WMMAs.
    // C layout: acc[v] is (row v + 8*hsel, col nt*16+rlo). ----
    const uint32_t sign_bit = 0x80000000u;
    for (int nt = 0; nt < NTILES; ++nt) {
        v8f acc0 = {}, acc1 = {};
        const uint32_t* bp = sB1 + nt * 256 + lane * 8;
#pragma unroll
        for (int ks = 0; ks < KSTEPS; ++ks) {
            FragU bf;
            const uint4* q = (const uint4*)(bp + ks * 2048);
            bf.u[0] = q[0];
            bf.u[1] = q[1];
            acc0 = __builtin_amdgcn_wmma_f32_16x16x32_f16(
                false, a1[0][ks], false, bf.h, (short)0, acc0, false, false);
            acc1 = __builtin_amdgcn_wmma_f32_16x16x32_f16(
                false, a1[1][ks], false, bf.h, (short)0, acc1, false, false);
        }
        const int colh  = nt * 16 + rlo;
        const int rbase = 8 * hsel;
#pragma unroll
        for (int c = 0; c < 2; ++c) {
#pragma unroll
            for (int v = 0; v < 8; ++v) {
                float val = (c == 0) ? acc0[v] : acc1[v];
                float a = __builtin_fabsf(val);
                float cc = qc[0];
#pragma unroll
                for (int i = 0; i < 7; ++i) cc = (a >= qb[i]) ? qc[i + 1] : cc;
                uint32_t sb = __float_as_uint(val) & sign_bit;
                float qv = __uint_as_float(__float_as_uint(cc) ^ sb);
                myS[(c * 16 + rbase + v) * D + colh] = (_Float16)qv;  // transpose via LDS
            }
        }
    }

    __asm__ __volatile__("s_wait_dscnt 0" ::: "memory");  // quant writes -> A2 reads

    // ---- A2 fragments (quantized values), fused rescale by fp16 norm. ----
    v16h a2[2][KSTEPS];
#pragma unroll
    for (int c = 0; c < 2; ++c) {
        _Float16 nh = (_Float16)((c == 0) ? nq0 : nq1);
        v16h nsplat;
#pragma unroll
        for (int i = 0; i < 16; ++i) nsplat[i] = nh;
#pragma unroll
        for (int ks = 0; ks < KSTEPS; ++ks) {
            FragU f;
            f.u[0] = *(const uint4*)(myS + c * 16 * D + hbase + ks * 32);
            f.u[1] = *(const uint4*)(myS + c * 16 * D + hbase + ks * 32 + 8);
            a2[c][ks] = f.h * nsplat;          // v_pk_mul_f16 x8
        }
    }

    // ---- GEMM2 (un-rotate), B2 streamed from L2-resident global frags,
    // each fragment load feeds 2 WMMAs. ----
    for (int nt = 0; nt < NTILES; ++nt) {
        v8f acc0 = {}, acc1 = {};
        const uint32_t* bp = b2g + nt * 256 + lane * 8;
#pragma unroll
        for (int ks = 0; ks < KSTEPS; ++ks) {
            FragU bf;
            const uint4* q = (const uint4*)(bp + ks * 2048);
            bf.u[0] = q[0];
            bf.u[1] = q[1];
            acc0 = __builtin_amdgcn_wmma_f32_16x16x32_f16(
                false, a2[0][ks], false, bf.h, (short)0, acc0, false, false);
            acc1 = __builtin_amdgcn_wmma_f32_16x16x32_f16(
                false, a2[1][ks], false, bf.h, (short)0, acc1, false, false);
        }
        const int col = nt * 16 + rlo;
#pragma unroll
        for (int v = 0; v < 8; ++v) {
            outr[(8 * hsel + v) * D + col] = acc0[v];
            outr[(16 + 8 * hsel + v) * D + col] = acc1[v];
        }
    }
}

// ---------------------------------------------------------------------------
extern "C" void kernel_launch(void* const* d_in, const int* in_sizes, int n_in,
                              void* d_out, int out_size, void* d_ws, size_t ws_size,
                              hipStream_t stream) {
    const float* x   = (const float*)d_in[0];
    const float* Pi  = (const float*)d_in[1];
    const float* cen = (const float*)d_in[2];
    const float* bnd = (const float*)d_in[3];
    float*       out = (float*)d_out;

    uint32_t* b1 = (uint32_t*)d_ws;                // 32 KB
    uint32_t* b2 = b1 + KSTEPS * NTILES * 32 * 8;  // next 32 KB

    const int nrows = in_sizes[0] / D;

    tq_prep_frags<<<32, 256, 0, stream>>>(Pi, b1, b2);

    const int rows_per_block = WAVES * ROWS_PER_WAVE;  // 128
    const int nblocks = (nrows + rows_per_block - 1) / rows_per_block;
    tq_main<<<nblocks, WAVES * 32, 0, stream>>>(x, cen, bnd, b1, b2, out, nrows);
}
